// GCNNMultiKernel_71305047048581
// MI455X (gfx1250) — compile-verified, matrix operand order checked by reading, not run on hardware
//
#include <hip/hip_runtime.h>
#include <hip/hip_bf16.h>

#define NN 2048
#define EE 4

typedef __attribute__((ext_vector_type(16))) __bf16 v16bf;
typedef __attribute__((ext_vector_type(2)))  __bf16 v2bf;
typedef __attribute__((ext_vector_type(8)))  float  v8f;

#ifdef __has_builtin
#if __has_builtin(__builtin_amdgcn_tensor_load_to_lds) && \
    __has_builtin(__builtin_amdgcn_s_wait_tensorcnt)
#define USE_TDM 1
typedef __attribute__((ext_vector_type(4))) unsigned int u32x4;
typedef __attribute__((ext_vector_type(4))) int          i32x4;
typedef __attribute__((ext_vector_type(8))) int          i32x8;
#endif
#endif

__device__ inline v16bf mk16(uint4 lo, uint4 hi) {
  union { unsigned int u[8]; v16bf v; } t;
  t.u[0] = lo.x; t.u[1] = lo.y; t.u[2] = lo.z; t.u[3] = lo.w;
  t.u[4] = hi.x; t.u[5] = hi.y; t.u[6] = hi.z; t.u[7] = hi.w;
  return t.v;
}
__device__ inline unsigned int f2bf2(float a, float b) {  // one v_cvt_pk_bf16_f32
  v2bf p = {(__bf16)a, (__bf16)b};
  return __builtin_bit_cast(unsigned int, p);
}
__device__ inline unsigned short f2bf(float x) {
  __bf16 h = (__bf16)x;
  return __builtin_bit_cast(unsigned short, h);
}
__device__ inline float bf2f(unsigned short u) {
  return (float)__builtin_bit_cast(__bf16, u);
}

// Fused: gaussian kernel construction + adj/deg contraction (WMMA over m) +
// spread assembly + (w_lin|w_nlin) head (WMMA over 9F) -> 32-ch embedding out.
// Block = 8 waves; each wave owns a 16-node output tile; block stages full
// per-batch embedding (bf16) + coords + bf16 weights in LDS.
template<int F>
__global__ __launch_bounds__(256)
void fused_gop(const float* __restrict__ coords_src,   // global_input (B,16,N)
               const float* __restrict__ emb_src,      // (B,F,N)
               const float* __restrict__ sigmas,       // E for this layer
               const float* __restrict__ w_lin,        // (16, 9F)
               const float* __restrict__ w_nlin,       // (16, 9F)
               const float* __restrict__ b_lin,        // (16)
               const float* __restrict__ b_nlin,       // (16)
               float* __restrict__ emb_out)            // (B,32,N)
{
  constexpr int NFT  = F / 16;                 // f-tiles of 16
  constexpr int K9   = 9 * F;                  // spread length
  constexpr int K9P  = (K9 + 31) & ~31;        // padded to WMMA K multiple
  constexpr int ESTR = NN + 8;                 // emb LDS row stride (u16)
  constexpr int WSTR = K9P + 8;                // W / S LDS row stride (u16)
  constexpr int SSTR = WSTR;

  extern __shared__ char smem[];
  float*          xL   = (float*)smem;                                     // N (x row)
  float*          yL   = xL + NN;                                          // N (y row)
  unsigned short* embL = (unsigned short*)(smem + 2 * NN * sizeof(float)); // F*ESTR
  unsigned short* wL   = embL + F * ESTR;                                  // 32*WSTR
  unsigned short* SL   = wL + 32 * WSTR;                                   // 8*16*SSTR
  float*          degL = (float*)(SL + 8 * 16 * SSTR);                     // 8*E*16

  const int b    = blockIdx.y;
  const int t    = threadIdx.x;
  const int wave = t >> 5;
  const int lane = t & 31;
  const int l16  = lane & 15;
  const int g    = lane >> 4;

  // ---------- Stage A: stage coords (TDM), bf16 emb, bf16 weights ----------
  const float* gx = coords_src + (size_t)b * 16 * NN;   // rows 0,1 = x,y

#ifdef USE_TDM
  if (t < 32) {   // wave 0 drives the Tensor Data Mover: 2x2048 f32 tile -> LDS 0
    unsigned long long ga = (unsigned long long)gx;
    u32x4 g0 = {1u,                                   // count=1 (valid D#)
                0u,                                   // lds_addr = 0 (xL base)
                (unsigned)(ga & 0xffffffffu),
                (unsigned)((ga >> 32) & 0x01ffffffu) | (2u << 30)}; // type=2
    i32x8 g1 = {(int)(2u << 16),        // data_size = 4 bytes
                (int)(2048u << 16),     // tensor_dim0 = 2048 (bits 79:48)
                (int)(2u << 16),        // tensor_dim1 = 2
                (int)(2048u << 16),     // tile_dim0 = 2048
                2,                      // tile_dim1 = 2, tile_dim2 = 0
                2048,                   // tensor_dim0_stride (low 32)
                0, 0};
    i32x4 g2 = {0, 0, 0, 0};
    i32x4 g3 = {0, 0, 0, 0};
#if __clang_major__ >= 23
    i32x8 g4 = {0, 0, 0, 0, 0, 0, 0, 0};
    __builtin_amdgcn_tensor_load_to_lds(g0, g1, g2, g3, g4, 0);
#else
    __builtin_amdgcn_tensor_load_to_lds(g0, g1, g2, g3, 0);
#endif
    __builtin_amdgcn_s_wait_tensorcnt(0);
  }
#else
  for (int m = t; m < NN; m += 256) { xL[m] = gx[m]; yL[m] = gx[NN + m]; }
#endif

  const float* es = emb_src + (size_t)b * F * NN;
  __builtin_prefetch(es + t * 16, 0, 0);
  __builtin_prefetch(w_lin + t, 0, 0);
  __builtin_prefetch(w_nlin + t, 0, 0);
  {
    const float4* es4 = (const float4*)es;
    for (int i4 = t; i4 < F * NN / 4; i4 += 256) {
      float4 v = es4[i4];
      int i = i4 * 4;
      int f = i >> 11, m = i & (NN - 1);
      uint2 pk;
      pk.x = f2bf2(v.x, v.y);
      pk.y = f2bf2(v.z, v.w);
      *(uint2*)(embL + f * ESTR + m) = pk;     // 8B-aligned ds_store_b64
    }
  }
  for (int i = t; i < 32 * WSTR; i += 256) {
    int o = i / WSTR, k = i - o * WSTR;
    float v = 0.f;
    if (k < K9) v = (o < 16) ? w_lin[o * K9 + k] : w_nlin[(o - 16) * K9 + k];
    wL[i] = f2bf(v);
  }
  __syncthreads();

  // ---------- Stage B: K_e tiles on the fly -> WMMA(adj) + WMMA(deg) ----------
  const int n0 = blockIdx.x * 128 + wave * 16;
  const float cnx = xL[n0 + l16];
  const float cny = yL[n0 + l16];

  float ce[EE];
#pragma unroll
  for (int e = 0; e < EE; ++e) {
    float s = sigmas[e];
    ce[e] = -1.4426950408889634f / (s * s);   // exp(-d2/s^2) = exp2(d2*ce)
  }

  const v8f vzero = {0.f, 0.f, 0.f, 0.f, 0.f, 0.f, 0.f, 0.f};
  v8f acc[EE][NFT + 1];
#pragma unroll
  for (int e = 0; e < EE; ++e)
#pragma unroll
    for (int ft = 0; ft <= NFT; ++ft) acc[e][ft] = vzero;

  v16bf bone;
  { uint4 q; q.x = q.y = q.z = q.w = 0x3f803f80u; bone = mk16(q, q); }

  for (int mb = 0; mb < NN; mb += 32) {
    // B tiles: emb[f][m] pairs, layout-matched b128 loads
    v16bf Bt[NFT];
#pragma unroll
    for (int ft = 0; ft < NFT; ++ft) {
      const unsigned short* p = embL + (ft * 16 + l16) * ESTR + mb + g * 8;
      Bt[ft] = mk16(*(const uint4*)p, *(const uint4*)(p + 16));
    }
    // d2 for this lane's 16 A slots: m = mb + 8g + idx (+8 for idx>=8)
    float d2[16];
    {
      const float* xr = xL + mb + g * 8;
      const float* yr = yL + mb + g * 8;
      float4 x0 = *(const float4*)(xr),      x1 = *(const float4*)(xr + 4);
      float4 x2 = *(const float4*)(xr + 16), x3 = *(const float4*)(xr + 20);
      float4 y0 = *(const float4*)(yr),      y1 = *(const float4*)(yr + 4);
      float4 y2 = *(const float4*)(yr + 16), y3 = *(const float4*)(yr + 20);
      float dx, dy;
      dx = cnx - x0.x; dy = cny - y0.x; d2[0]  = dx * dx + dy * dy;
      dx = cnx - x0.y; dy = cny - y0.y; d2[1]  = dx * dx + dy * dy;
      dx = cnx - x0.z; dy = cny - y0.z; d2[2]  = dx * dx + dy * dy;
      dx = cnx - x0.w; dy = cny - y0.w; d2[3]  = dx * dx + dy * dy;
      dx = cnx - x1.x; dy = cny - y1.x; d2[4]  = dx * dx + dy * dy;
      dx = cnx - x1.y; dy = cny - y1.y; d2[5]  = dx * dx + dy * dy;
      dx = cnx - x1.z; dy = cny - y1.z; d2[6]  = dx * dx + dy * dy;
      dx = cnx - x1.w; dy = cny - y1.w; d2[7]  = dx * dx + dy * dy;
      dx = cnx - x2.x; dy = cny - y2.x; d2[8]  = dx * dx + dy * dy;
      dx = cnx - x2.y; dy = cny - y2.y; d2[9]  = dx * dx + dy * dy;
      dx = cnx - x2.z; dy = cny - y2.z; d2[10] = dx * dx + dy * dy;
      dx = cnx - x2.w; dy = cny - y2.w; d2[11] = dx * dx + dy * dy;
      dx = cnx - x3.x; dy = cny - y3.x; d2[12] = dx * dx + dy * dy;
      dx = cnx - x3.y; dy = cny - y3.y; d2[13] = dx * dx + dy * dy;
      dx = cnx - x3.z; dy = cny - y3.z; d2[14] = dx * dx + dy * dy;
      dx = cnx - x3.w; dy = cny - y3.w; d2[15] = dx * dx + dy * dy;
    }
#pragma unroll
    for (int e = 0; e < EE; ++e) {
      unsigned int aw[8];
#pragma unroll
      for (int j = 0; j < 8; ++j) {
        float k0 = __builtin_amdgcn_exp2f(d2[2 * j]     * ce[e]);
        float k1 = __builtin_amdgcn_exp2f(d2[2 * j + 1] * ce[e]);
        aw[j] = f2bf2(k0, k1);               // single v_cvt_pk_bf16_f32
      }
      uint4 alo, ahi;
      alo.x = aw[0]; alo.y = aw[1]; alo.z = aw[2]; alo.w = aw[3];
      ahi.x = aw[4]; ahi.y = aw[5]; ahi.z = aw[6]; ahi.w = aw[7];
      v16bf Ae = mk16(alo, ahi);
#pragma unroll
      for (int ft = 0; ft < NFT; ++ft)
        acc[e][ft] = __builtin_amdgcn_wmma_f32_16x16x32_bf16(
            false, Ae, false, Bt[ft], (short)0, acc[e][ft], false, false);
      acc[e][NFT] = __builtin_amdgcn_wmma_f32_16x16x32_bf16(
          false, Ae, false, bone, (short)0, acc[e][NFT], false, false);
    }
  }

  // ---------- Stage C: build spread tile S (16 nodes x K9) in LDS ----------
  unsigned short* Sw = SL + wave * 16 * SSTR;
  float* dg = degL + wave * EE * 16;

  if (l16 == 0) {
#pragma unroll
    for (int e = 0; e < EE; ++e)
#pragma unroll
      for (int r = 0; r < 8; ++r)
        dg[e * 16 + 8 * g + r] = acc[e][NFT][r];      // deg[e, n=r+8g]
  }
  // adj rows: k = F + E*F + e*F + f
#pragma unroll
  for (int e = 0; e < EE; ++e)
#pragma unroll
    for (int ft = 0; ft < NFT; ++ft)
#pragma unroll
      for (int r = 0; r < 8; ++r)
        Sw[(r + 8 * g) * SSTR + (F + EE * F + e * F + ft * 16 + l16)] =
            f2bf(acc[e][ft][r]);
  // emb rows: k = f
#pragma unroll
  for (int j = 0; j < F / 2; ++j) {
    int f = g * (F / 2) + j;
    Sw[l16 * SSTR + f] = embL[f * ESTR + n0 + l16];
  }
  // zero K padding (F==16 case)
  if (K9P > K9) {
#pragma unroll
    for (int j = 0; j < (K9P - K9 + 1) / 2; ++j)
      Sw[l16 * SSTR + K9 + g * ((K9P - K9) / 2) + j] = 0;
  }
  __syncthreads();
  // deg_spread rows: k = F + e*F + f, value deg[e,n]*emb[f,n]
#pragma unroll
  for (int e = 0; e < EE; ++e) {
    float dv = dg[e * 16 + l16];
#pragma unroll
    for (int j = 0; j < F / 2; ++j) {
      int f = g * (F / 2) + j;
      float ev = bf2f(embL[f * ESTR + n0 + l16]);
      Sw[l16 * SSTR + (F + e * F + f)] = f2bf(dv * ev);
    }
  }
  __syncthreads();

  // ---------- Stage D: head = [w_lin; w_nlin](32 x K9) @ S(K9 x 16) ----------
  v8f oacc[2];
  oacc[0] = vzero; oacc[1] = vzero;
  for (int kc = 0; kc < K9P; kc += 32) {
    const unsigned short* sp = Sw + l16 * SSTR + kc + g * 8;
    v16bf Bs = mk16(*(const uint4*)sp, *(const uint4*)(sp + 16));
#pragma unroll
    for (int ot = 0; ot < 2; ++ot) {
      const unsigned short* wp = wL + (ot * 16 + l16) * WSTR + kc + g * 8;
      v16bf Aw = mk16(*(const uint4*)wp, *(const uint4*)(wp + 16));
      oacc[ot] = __builtin_amdgcn_wmma_f32_16x16x32_bf16(
          false, Aw, false, Bs, (short)0, oacc[ot], false, false);
    }
  }

  const int n = n0 + l16;
  float* outp = emb_out + (size_t)b * 32 * NN;
#pragma unroll
  for (int r = 0; r < 8; ++r) {
    int o = r + 8 * g;
    outp[o * NN + n]        = oacc[0][r] + b_lin[o];                // lin
    float v1 = oacc[1][r] + b_nlin[o];
    outp[(16 + o) * NN + n] = fmaxf(v1, 0.f);                       // relu
  }
}

// spatialnorm over nodes: (x - mean) / (std_ddof1 + 1e-8)
__global__ __launch_bounds__(256)
void spatial_norm(const float* __restrict__ in, float* __restrict__ out) {
  const int f = blockIdx.x, b = blockIdx.y;
  const float* p = in  + ((size_t)b * 32 + f) * NN;
  float*       q = out + ((size_t)b * 32 + f) * NN;
  __shared__ float red[256];
  const int t = threadIdx.x;
  float s = 0.f;
  for (int i = t; i < NN; i += 256) s += p[i];
  red[t] = s; __syncthreads();
  for (int w = 128; w > 0; w >>= 1) { if (t < w) red[t] += red[t + w]; __syncthreads(); }
  float mu = red[0] / (float)NN;
  __syncthreads();
  float ss = 0.f;
  for (int i = t; i < NN; i += 256) { float d = p[i] - mu; ss += d * d; }
  red[t] = ss; __syncthreads();
  for (int w = 128; w > 0; w >>= 1) { if (t < w) red[t] += red[t + w]; __syncthreads(); }
  float sd  = sqrtf(red[0] / (float)(NN - 1));
  float inv = 1.f / (sd + 1e-8f);
  for (int i = t; i < NN; i += 256) q[i] = (p[i] - mu) * inv;
}

// pooled mean -> instance norm over 32 ch -> fcl -> sigmoid
__global__ void final_head(const float* __restrict__ emb,
                           const float* __restrict__ fw,
                           const float* __restrict__ fb,
                           float* __restrict__ out) {
  const int b = blockIdx.x;
  const int o = threadIdx.x;                  // 32 lanes = 32 channels
  const float* p = emb + ((size_t)b * 32 + o) * NN;
  float s = 0.f;
  for (int i = 0; i < NN; ++i) s += p[i];
  float pooled = s / (float)NN;
  float m = pooled;
  for (int d = 16; d > 0; d >>= 1) m += __shfl_xor(m, d, 32);
  m *= (1.f / 32.f);
  float dv = pooled - m;
  float v = dv * dv;
  for (int d = 16; d > 0; d >>= 1) v += __shfl_xor(v, d, 32);
  v *= (1.f / 32.f);
  float normed = dv * rsqrtf(v + 1e-5f);
  float c = normed * fw[o];
  for (int d = 16; d > 0; d >>= 1) c += __shfl_xor(c, d, 32);
  if (o == 0) out[b] = 1.f / (1.f + expf(-(c + fb[0])));
}

static size_t smem_for(int F) {
  int K9 = 9 * F, K9P = (K9 + 31) & ~31;
  int ESTR = NN + 8, WSTR = K9P + 8;
  return (size_t)2 * NN * 4                 // coords (x row, y row)
       + (size_t)F * ESTR * 2               // emb bf16
       + (size_t)32 * WSTR * 2              // weights bf16
       + (size_t)8 * 16 * WSTR * 2          // per-wave spread tiles
       + (size_t)8 * EE * 16 * 4;           // deg scratch
}

extern "C" void kernel_launch(void* const* d_in, const int* in_sizes, int n_in,
                              void* d_out, int out_size, void* d_ws, size_t ws_size,
                              hipStream_t stream) {
  (void)in_sizes; (void)n_in; (void)out_size; (void)ws_size;
  const float* gi  = (const float*)d_in[0];
  const float* sig = (const float*)d_in[1];
  const float* wl0 = (const float*)d_in[2];
  const float* bl0 = (const float*)d_in[3];
  const float* wn0 = (const float*)d_in[4];
  const float* bn0 = (const float*)d_in[5];
  const float* wl  = (const float*)d_in[6];
  const float* bl  = (const float*)d_in[7];
  const float* wn  = (const float*)d_in[8];
  const float* bn  = (const float*)d_in[9];
  const float* fw  = (const float*)d_in[10];
  const float* fb  = (const float*)d_in[11];
  float* out = (float*)d_out;

  float* bufA = (float*)d_ws;                 // (B,32,N)
  float* bufB = bufA + (size_t)4 * 32 * NN;   // (B,32,N)

  dim3 grid(NN / 128, 4), blk(256);

  // Layer 0: emb = global_input (F=16)
  fused_gop<16><<<grid, blk, smem_for(16), stream>>>(
      gi, gi, sig, wl0, wn0, bl0, bn0, bufA);

  // Layers 1..2: spatialnorm then fused conv (F=32)
  for (int L = 0; L < 2; ++L) {
    spatial_norm<<<dim3(32, 4), 256, 0, stream>>>(bufA, bufB);
    fused_gop<32><<<grid, blk, smem_for(32), stream>>>(
        gi, bufB, sig + (L + 1) * EE,
        wl + (size_t)L * 16 * 288, wn + (size_t)L * 16 * 288,
        bl + L * 16, bn + L * 16, bufA);
  }

  final_head<<<dim3(4), dim3(32), 0, stream>>>(bufA, fw, fb, out);
}
